// MPNN_14705968022404
// MI455X (gfx1250) — matrix operand order repaired and checked by reading, not compiled, not run on hardware
//
#include <hip/hip_runtime.h>
#include <hip/hip_bf16.h>
#include <math.h>

typedef __bf16 bf16_t;
typedef __attribute__((ext_vector_type(16))) __bf16 v16bf;
typedef __attribute__((ext_vector_type(8)))  float  v8f;

#define N_NODES 3
#define BDIM    4
#define NB      12          // N_NODES * BDIM
#define C_DIM   256
#define C2      512
#define H_DIM   40
#define W_DIM   40
#define P_DIM   1600
#define HP      42          // padded height (1 halo row each side)
#define WP      48          // padded width  (1 halo col + pad to 48)

union FragBF { v16bf v; uint32_t u[8]; };

__device__ __forceinline__ bf16_t f2bf(float x) { return (bf16_t)x; }

// ---- CDNA5 async global->LDS staging (ASYNCcnt path, CDNA5 ISA) ----
// GVS mode: LDS address comes from the vdst VGPR; global address = SGPR64 base + VGPR32 voffset.
__device__ __forceinline__ void async_b128(const bf16_t* lds_dst, const void* sbase,
                                           uint32_t voff_bytes) {
  uint32_t ldso = (uint32_t)(uintptr_t)lds_dst;   // low 32 bits = LDS offset
  asm volatile("global_load_async_to_lds_b128 %0, %1, %2"
               :: "v"(ldso), "v"(voff_bytes), "s"(sbase)
               : "memory");
}
// Partial drain: wait until wave's outstanding async transfers <= N.
// (async loads signal done in order -> count-based pipelining is safe)
template <int N>
__device__ __forceinline__ void wait_async() {
  asm volatile("s_wait_asynccnt %0" :: "i"(N) : "memory");
}

// A fragment (16x32 bf16), [M][K] row-major (LDS or global), stride strideK elems.
// ISA 7.12.2: lane<16 holds K pairs 0..7 & 16..23; lane>=16 holds 8..15 & 24..31.
__device__ __forceinline__ v16bf matA_frag(const bf16_t* base, int strideK) {
  const int l = threadIdx.x & 31;
  const int r = l & 15, kh = l >> 4;
  const bf16_t* p = base + r * strideK + kh * 8;
  FragBF f;
#pragma unroll
  for (int i = 0; i < 4; ++i) f.u[i]     = *(const uint32_t*)(p + 2 * i);
#pragma unroll
  for (int i = 0; i < 4; ++i) f.u[4 + i] = *(const uint32_t*)(p + 16 + 2 * i);
  return f.v;
}

// B fragment (32x16 bf16), LDS [N][K] layout (pairs along K contiguous -> b128).
__device__ __forceinline__ v16bf ldsB_frag_nk(const bf16_t* base, int strideN) {
  const int l = threadIdx.x & 31;
  const int n = l & 15, kh = l >> 4;
  const bf16_t* p = base + n * strideN + kh * 8;
  FragBF f;
#pragma unroll
  for (int i = 0; i < 4; ++i) f.u[i]     = *(const uint32_t*)(p + 2 * i);
#pragma unroll
  for (int i = 0; i < 4; ++i) f.u[4 + i] = *(const uint32_t*)(p + 16 + 2 * i);
  return f.v;
}

__device__ __forceinline__ v8f wmma_bf16(v16bf a, v16bf b, v8f c) {
  return __builtin_amdgcn_wmma_f32_16x16x32_bf16(false, a, false, b, (short)0, c,
                                                 false, false);
}

// ---------------- elementwise helpers ----------------

__global__ void f2bf_kernel(const float* __restrict__ s, bf16_t* __restrict__ d, int n) {
  int i = blockIdx.x * blockDim.x + threadIdx.x;
  if (i < n) d[i] = f2bf(s[i]);
}

// NCHW [nb][c][p] -> channels-last [nb][p][c] (fp32)
__global__ void to_cl_kernel(const float* __restrict__ s, float* __restrict__ d) {
  int i = blockIdx.x * blockDim.x + threadIdx.x;
  const int total = NB * C_DIM * P_DIM;
  if (i >= total) return;
  int c  = i % C_DIM;
  int p  = (i / C_DIM) % P_DIM;
  int nb = i / (C_DIM * P_DIM);
  d[i] = s[((size_t)nb * C_DIM + c) * P_DIM + p];
}

// channels-last [nb][p][c] -> NCHW [nb][c][p] (fp32)
__global__ void from_cl_kernel(const float* __restrict__ s, float* __restrict__ d) {
  int i = blockIdx.x * blockDim.x + threadIdx.x;
  const int total = NB * C_DIM * P_DIM;
  if (i >= total) return;
  int p  = i % P_DIM;
  int c  = (i / P_DIM) % C_DIM;
  int nb = i / (C_DIM * P_DIM);
  d[i] = s[((size_t)nb * P_DIM + p) * C_DIM + c];
}

// Pair sums for attention V: vs[((pair*B+b)*C + c)*P + q] = bf16(h_i + h_j)
// pair 0={0,1}, 1={0,2}, 2={1,2}; h is channels-last [nb][p][c] fp32.
__global__ void vsum_kernel(const float* __restrict__ ncl, bf16_t* __restrict__ vs) {
  int i = blockIdx.x * blockDim.x + threadIdx.x;
  const int total = 3 * BDIM * C_DIM * P_DIM;
  if (i >= total) return;
  int q    = i % P_DIM;
  int c    = (i / P_DIM) % C_DIM;
  int pb   = i / (P_DIM * C_DIM);
  int b    = pb % BDIM;
  int pair = pb / BDIM;
  int n1 = (pair == 2) ? 1 : 0;
  int n2 = (pair == 0) ? 1 : 2;
  float v = ncl[((size_t)(n1 * BDIM + b) * P_DIM + q) * C_DIM + c] +
            ncl[((size_t)(n2 * BDIM + b) * P_DIM + q) * C_DIM + c];
  vs[i] = f2bf(v);
}

// w[o][ci][3][3] (fp32) -> wt[tap][o][ci] (bf16), K-contiguous rows
__global__ void wtrans_kernel(const float* __restrict__ w, bf16_t* __restrict__ wt, int Cout) {
  int i = blockIdx.x * blockDim.x + threadIdx.x;
  int total = Cout * C2 * 9;
  if (i >= total) return;
  int ci  = i % C2;
  int o   = (i / C2) % Cout;
  int tap = i / (C2 * Cout);
  wt[i] = f2bf(w[((size_t)o * C2 + ci) * 9 + tap]);
}

// Zero-padded bf16 concat([srcA, srcB]) channels-last: [NB][42][48][512]
__global__ void build_pad_kernel(const float* __restrict__ srcA,
                                 const float* __restrict__ srcB,
                                 bf16_t* __restrict__ dst) {
  int i = blockIdx.x * blockDim.x + threadIdx.x;
  const int total = NB * HP * WP * C2;
  if (i >= total) return;
  int cc = i % C2;
  int x  = (i / C2) % WP;
  int r  = (i / (C2 * WP)) % HP;
  int nb = i / (C2 * WP * HP);
  float v = 0.f;
  int y = r - 1, xx = x - 1;
  if (y >= 0 && y < H_DIM && xx >= 0 && xx < W_DIM) {
    const float* s = (cc < C_DIM)
        ? srcA + ((size_t)nb * P_DIM + (y * W_DIM + xx)) * C_DIM + cc
        : srcB + ((size_t)nb * P_DIM + (y * W_DIM + xx)) * C_DIM + (cc - C_DIM);
    v = *s;
  }
  dst[i] = f2bf(v);
}

// ---------------- theta: 1x1 conv == 256x256 @ 256x1600 GEMM ----------------
// grid (10, 2, NB), block 256 (8 waves). A-frags from global (weights are
// L2-resident); B tiles double-buffered in LDS with async prefetch.
__global__ __launch_bounds__(256)
void theta_kernel(const bf16_t* __restrict__ twbf, const float* __restrict__ tb,
                  const bf16_t* __restrict__ ntbf, bf16_t* __restrict__ thbf) {
  __shared__ bf16_t ldsB[2][160 * 40];   // [p][c] ([N][K]), stride 40
  const int nb  = blockIdx.z;
  const int m0  = blockIdx.y * 128;
  const int p0  = blockIdx.x * 160;
  const int tid = threadIdx.x;
  const int wave = tid >> 5;
  const int l = tid & 31;
  const int n = l & 15, rbm = (l >> 4) * 8;
  const bf16_t* baseB = ntbf + ((size_t)nb * P_DIM + p0) * C_DIM;

  v8f acc[10];
#pragma unroll
  for (int j = 0; j < 10; ++j)
#pragma unroll
    for (int i = 0; i < 8; ++i) acc[j][i] = 0.f;

  // prologue: prefetch B chunk 0
  for (int idx = tid; idx < 160 * 4; idx += 256) {
    int p = idx >> 2, seg = idx & 3;
    async_b128(&ldsB[0][p * 40 + seg * 8], baseB, (uint32_t)(p * 512 + seg * 16));
  }
  for (int kci = 0; kci < 8; ++kci) {
    const int kc = kci * 32;
    const int cur = kci & 1;
    if (kci < 7) {   // prefetch next chunk into other buffer
      for (int idx = tid; idx < 160 * 4; idx += 256) {
        int p = idx >> 2, seg = idx & 3;
        async_b128(&ldsB[cur ^ 1][p * 40 + seg * 8], baseB + kc + 32,
                   (uint32_t)(p * 512 + seg * 16));
      }
      wait_async<2>();   // drains chunk kci (conservative across waves)
    } else {
      wait_async<0>();
    }
    __syncthreads();
    v16bf a = matA_frag(twbf + (size_t)(m0 + wave * 16) * C_DIM + kc, C_DIM);
    // batch all B-frag loads, then run WMMAs back-to-back (one dscnt wait)
    v16bf bfr[10];
#pragma unroll
    for (int j = 0; j < 10; ++j) bfr[j] = ldsB_frag_nk(&ldsB[cur][j * 16 * 40], 40);
#pragma unroll
    for (int j = 0; j < 10; ++j) acc[j] = wmma_bf16(a, bfr[j], acc[j]);
    __syncthreads();   // everyone done with buf before it is refilled
  }
#pragma unroll
  for (int j = 0; j < 10; ++j) {
#pragma unroll
    for (int i = 0; i < 8; ++i) {
      int o = m0 + wave * 16 + rbm + i;
      int p = p0 + j * 16 + n;
      thbf[((size_t)nb * P_DIM + p) * C_DIM + o] = f2bf(acc[j][i] + tb[o]);
    }
  }
}

// ---------------- flash attention + segment-sum ----------------
// grid (25, NB), block 128 (4 waves). Q in registers; K double-buffered with
// async prefetch; V async-loaded under S-compute. 3 barriers per q-chunk.
__global__ __launch_bounds__(128)
void attn_kernel(const bf16_t* __restrict__ thbf, const bf16_t* __restrict__ ntbf,
                 const bf16_t* __restrict__ vsum, float* __restrict__ agg) {
  __shared__ bf16_t ldsK[2][32 * 264];  // [q][c] stride 264, double buffer (33.8KB)
  __shared__ bf16_t ldsV[256 * 40];     // [c][q] stride 40 (20.5KB)
  __shared__ bf16_t ldsP[4 * 16 * 40];  // per-wave P tile (5.1KB)
  const int nb    = blockIdx.y;
  const int nnode = nb / BDIM;
  const int b     = nb % BDIM;
  const int p0    = blockIdx.x * 64;
  const int tid   = threadIdx.x;
  const int wave  = tid >> 5;
  const int l     = tid & 31;
  const int n     = l & 15, rb = (l >> 4) * 8;

  // Q = theta(h_recv)^T in registers: 8 A-frags per wave, from global [p][c]
  v16bf qf[8];
  {
    const bf16_t* qbase = thbf + ((size_t)nb * P_DIM + p0 + wave * 16) * C_DIM;
#pragma unroll
    for (int kc = 0; kc < 8; ++kc) qf[kc] = matA_frag(qbase + kc * 32, C_DIM);
  }

  const int s0n = (nnode == 0) ? 1 : 0;   // the two sender nodes != nnode
  const int s1n = (nnode == 2) ? 1 : 2;

  for (int je = 0; je < 2; ++je) {
    const int snode = (je == 0) ? s0n : s1n;
    const int snb   = snode * BDIM + b;
    const int pair  = nnode + snode - 1;   // {i,j} -> i+j-1
    const bf16_t* baseK = ntbf + (size_t)snb * P_DIM * C_DIM;
    const bf16_t* baseV = vsum + ((size_t)(pair * BDIM + b) * C_DIM) * P_DIM;

    float m[8], lsum[8];
#pragma unroll
    for (int i = 0; i < 8; ++i) { m[i] = -3.0e38f; lsum[i] = 0.f; }
    v8f O[16];
#pragma unroll
    for (int cn = 0; cn < 16; ++cn)
#pragma unroll
      for (int i = 0; i < 8; ++i) O[cn][i] = 0.f;

    // prologue: K chunk 0 into buffer 0 (8 async b128 per thread)
    for (int idx = tid; idx < 1024; idx += 128) {
      int q = idx >> 5, seg = idx & 31;
      async_b128(&ldsK[0][q * 264 + seg * 8], baseK, (uint32_t)(q * 512 + seg * 16));
    }

    for (int it = 0; it < 50; ++it) {
      const int q0 = it * 32;
      const int cur = it & 1;
      // issue V[it] (overlaps with S compute)
      for (int idx = tid; idx < 1024; idx += 128) {
        int c = idx >> 2, seg = idx & 3;
        async_b128(&ldsV[c * 40 + seg * 8], baseV + q0,
                   (uint32_t)(c * (P_DIM * 2) + seg * 16));
      }
      if (it < 49) {   // prefetch K[it+1] into the other buffer
        for (int idx = tid; idx < 1024; idx += 128) {
          int q = idx >> 5, seg = idx & 31;
          async_b128(&ldsK[cur ^ 1][q * 264 + seg * 8],
                     baseK + (size_t)(q0 + 32) * C_DIM,
                     (uint32_t)(q * 512 + seg * 16));
        }
        wait_async<16>();   // V[it](8) + K[it+1](8) in flight; K[it] drained
      } else {
        wait_async<8>();    // V[it](8) in flight; K[it] drained
      }
      __syncthreads();

      v8f S0, S1;
#pragma unroll
      for (int i = 0; i < 8; ++i) { S0[i] = 0.f; S1[i] = 0.f; }
      // kc pairs: 4 frag loads batched, then 4 WMMAs (2 independent chains)
#pragma unroll
      for (int kc = 0; kc < 8; kc += 2) {
        v16bf b00 = ldsB_frag_nk(&ldsK[cur][kc * 32], 264);
        v16bf b10 = ldsB_frag_nk(&ldsK[cur][16 * 264 + kc * 32], 264);
        v16bf b01 = ldsB_frag_nk(&ldsK[cur][(kc + 1) * 32], 264);
        v16bf b11 = ldsB_frag_nk(&ldsK[cur][16 * 264 + (kc + 1) * 32], 264);
        S0 = wmma_bf16(qf[kc], b00, S0);
        S1 = wmma_bf16(qf[kc], b10, S1);
        S0 = wmma_bf16(qf[kc + 1], b01, S0);
        S1 = wmma_bf16(qf[kc + 1], b11, S1);
      }
      // online softmax: rows in 16-lane groups (C/D layout), shfl_xor reductions
      float alpha[8];
#pragma unroll
      for (int i = 0; i < 8; ++i) {
        float t = fmaxf(S0[i], S1[i]);
#pragma unroll
        for (int msk = 8; msk >= 1; msk >>= 1) t = fmaxf(t, __shfl_xor(t, msk, 32));
        float mn  = fmaxf(m[i], t);
        float p0e = __expf(S0[i] - mn);
        float p1e = __expf(S1[i] - mn);
        float rs  = p0e + p1e;
#pragma unroll
        for (int msk = 8; msk >= 1; msk >>= 1) rs += __shfl_xor(rs, msk, 32);
        alpha[i] = __expf(m[i] - mn);
        lsum[i]  = lsum[i] * alpha[i] + rs;
        m[i] = mn;
        ldsP[wave * 640 + (rb + i) * 40 + n]      = f2bf(p0e);
        ldsP[wave * 640 + (rb + i) * 40 + 16 + n] = f2bf(p1e);
      }
#pragma unroll
      for (int cn = 0; cn < 16; ++cn)
#pragma unroll
        for (int i = 0; i < 8; ++i) O[cn][i] *= alpha[i];

      if (it < 49) wait_async<8>();  // V[it] landed (K[it+1] still in flight)
      else         wait_async<0>();
      __syncthreads();

      v16bf aP = matA_frag(ldsP + wave * 640, 40);
      // PV in groups of 4: batch frag loads, then 4 independent WMMA chains
#pragma unroll
      for (int cng = 0; cng < 16; cng += 4) {
        v16bf bv[4];
#pragma unroll
        for (int k = 0; k < 4; ++k)
          bv[k] = ldsB_frag_nk(&ldsV[(cng + k) * 16 * 40], 40);
#pragma unroll
        for (int k = 0; k < 4; ++k)
          O[cng + k] = wmma_bf16(aP, bv[k], O[cng + k]);
      }
      __syncthreads();   // protect ldsV before next V issue
    }
    // epilogue: normalize; segment-sum folded in (edge0 store, edge1 add)
    // agg is [nb][p][c]: 16 lanes cover 64B contiguous per (cn,i)
#pragma unroll
    for (int cn = 0; cn < 16; ++cn) {
#pragma unroll
      for (int i = 0; i < 8; ++i) {
        int prow = p0 + wave * 16 + rb + i;
        int c    = cn * 16 + n;
        size_t gi = ((size_t)nb * P_DIM + prow) * C_DIM + c;
        float val = O[cn][i] / lsum[i];
        if (je == 0) agg[gi] = val;
        else         agg[gi] += val;
      }
    }
  }
}

// ---------------- conv3x3 implicit GEMM + fused GRU epilogue ----------------
// grid (10, COUT/128, NB), block 256 (8 waves). Weight A-frags from global
// (L2-resident); activation tiles double-buffered with async prefetch.
template <int COUT, int MODE>
__global__ __launch_bounds__(256)
void conv3x3_kernel(const bf16_t* __restrict__ xpad, const bf16_t* __restrict__ wt,
                    const float* __restrict__ bias, const float* __restrict__ h,
                    const float* __restrict__ zin, float* __restrict__ out0,
                    float* __restrict__ out1) {
  __shared__ bf16_t ldsAct[2][290 * 40 + 16];  // [s][ci], double buffer (46.5KB)
  const int nb  = blockIdx.z;
  const int m0  = blockIdx.y * 128;
  const int y0  = blockIdx.x * 4;
  const int tid = threadIdx.x;
  const int wave = tid >> 5;
  const int l = tid & 31;
  const int n = l & 15, rbm = (l >> 4) * 8;
  const bf16_t* baseAct = xpad + ((size_t)(nb * HP + y0) * WP) * C2;

  v8f acc[12];
#pragma unroll
  for (int t = 0; t < 12; ++t)
#pragma unroll
    for (int i = 0; i < 8; ++i) acc[t][i] = 0.f;

  // prologue: activation chunk 0 (288 spatial x 64B)
  for (int idx = tid; idx < 288 * 4; idx += 256) {
    int s = idx >> 2, seg = idx & 3;
    async_b128(&ldsAct[0][s * 40 + seg * 8], baseAct, (uint32_t)(s * 1024 + seg * 16));
  }
  for (int cci = 0; cci < 16; ++cci) {
    const int cc = cci * 32;
    const int cur = cci & 1;
    if (cci < 15) {   // prefetch next cin-chunk
      for (int idx = tid; idx < 288 * 4; idx += 256) {
        int s = idx >> 2, seg = idx & 3;
        async_b128(&ldsAct[cur ^ 1][s * 40 + seg * 8], baseAct + cc + 32,
                   (uint32_t)(s * 1024 + seg * 16));
      }
      wait_async<4>();   // drains chunk cci (conservative across waves)
    } else {
      wait_async<0>();
    }
    __syncthreads();
    for (int tap = 0; tap < 9; ++tap) {
      const int dy = tap / 3, dx = tap % 3;
      v16bf a = matA_frag(wt + ((size_t)tap * COUT + m0 + wave * 16) * C2 + cc, C2);
      // batch all 12 B-frag loads, then 12 independent WMMA chains
      v16bf bfr[12];
#pragma unroll
      for (int yl = 0; yl < 4; ++yl)
#pragma unroll
        for (int j = 0; j < 3; ++j) {
          int sbase = (yl + dy) * WP + j * 16 + dx;
          bfr[yl * 3 + j] = ldsB_frag_nk(&ldsAct[cur][sbase * 40], 40);
        }
#pragma unroll
      for (int t = 0; t < 12; ++t) acc[t] = wmma_bf16(a, bfr[t], acc[t]);
    }
    __syncthreads();   // everyone done with buf before it is refilled
  }
  // fused epilogue (channels-last: o consecutive per lane -> contiguous LD/ST)
#pragma unroll
  for (int yl = 0; yl < 4; ++yl) {
#pragma unroll
    for (int j = 0; j < 3; ++j) {
      int x = j * 16 + n;
      if (x >= W_DIM) continue;
#pragma unroll
      for (int i = 0; i < 8; ++i) {
        int o = m0 + wave * 16 + rbm + i;
        int p = (y0 + yl) * W_DIM + x;
        float val = acc[yl * 3 + j][i] + bias[o];
        if (MODE == 0) {                      // gates: sigmoid -> r*h and z
          float s = 1.f / (1.f + __expf(-val));
          if (o < C_DIM) {
            size_t gi = ((size_t)nb * P_DIM + p) * C_DIM + o;
            out0[gi] = s * h[gi];             // rh
          } else {
            size_t gi = ((size_t)nb * P_DIM + p) * C_DIM + (o - C_DIM);
            out1[gi] = s;                     // z
          }
        } else {                              // cand: tanh + GRU update in place
          size_t gi = ((size_t)nb * P_DIM + p) * C_DIM + o;
          float cd = tanhf(val);
          float zv = zin[gi];
          out0[gi] = (1.f - zv) * h[gi] + zv * cd;
        }
      }
    }
  }
}

// ---------------- launcher ----------------
extern "C" void kernel_launch(void* const* d_in, const int* in_sizes, int n_in,
                              void* d_out, int out_size, void* d_ws, size_t ws_size,
                              hipStream_t stream) {
  (void)in_sizes; (void)n_in; (void)out_size; (void)ws_size;
  const float* in_nodes = (const float*)d_in[0];
  const float* theta_w  = (const float*)d_in[1];
  const float* theta_b  = (const float*)d_in[2];
  const float* gates_w  = (const float*)d_in[3];
  const float* gates_b  = (const float*)d_in[4];
  const float* can_w    = (const float*)d_in[5];
  const float* can_b    = (const float*)d_in[6];

  char* ws = (char*)d_ws;
  size_t off = 0;
  auto carve = [&](size_t bytes) {
    char* p = ws + off;
    off += (bytes + 255) & ~(size_t)255;
    return p;
  };
  const size_t NBCP = (size_t)NB * C_DIM * P_DIM;
  float*  nodes_cl = (float*)carve(NBCP * 4);      // h state, [nb][p][c]
  float*  agg      = (float*)carve(NBCP * 4);      // [nb][p][c]
  float*  zbuf     = (float*)carve(NBCP * 4);      // [nb][p][c]
  float*  rh       = (float*)carve(NBCP * 4);      // [nb][p][c]
  bf16_t* thbf     = (bf16_t*)carve(NBCP * 2);     // [nb][p][c]
  bf16_t* ntbf     = (bf16_t*)carve(NBCP * 2);     // [nb][p][c]
  bf16_t* vsum     = (bf16_t*)carve((size_t)3 * BDIM * C_DIM * P_DIM * 2); // [pair][b][c][p]
  bf16_t* xpad     = (bf16_t*)carve((size_t)NB * HP * WP * C2 * 2);
  bf16_t* wg_t     = (bf16_t*)carve((size_t)C2 * C2 * 9 * 2);
  bf16_t* wc_t     = (bf16_t*)carve((size_t)C_DIM * C2 * 9 * 2);
  bf16_t* twbf     = (bf16_t*)carve((size_t)C_DIM * C_DIM * 2);

  const int ETOT = (int)NBCP;
  to_cl_kernel<<<(ETOT + 255) / 256, 256, 0, stream>>>(in_nodes, nodes_cl);
  {
    int t = C_DIM * C_DIM;
    f2bf_kernel<<<(t + 255) / 256, 256, 0, stream>>>(theta_w, twbf, t);
    int tg = C2 * C2 * 9;
    wtrans_kernel<<<(tg + 255) / 256, 256, 0, stream>>>(gates_w, wg_t, C2);
    int tc = C_DIM * C2 * 9;
    wtrans_kernel<<<(tc + 255) / 256, 256, 0, stream>>>(can_w, wc_t, C_DIM);
  }

  for (int pass = 0; pass < 2; ++pass) {
    f2bf_kernel<<<(ETOT + 255) / 256, 256, 0, stream>>>(nodes_cl, ntbf, ETOT);
    theta_kernel<<<dim3(10, 2, NB), 256, 0, stream>>>(twbf, theta_b, ntbf, thbf);
    vsum_kernel<<<(ETOT + 255) / 256, 256, 0, stream>>>(nodes_cl, vsum);
    attn_kernel<<<dim3(25, NB), 128, 0, stream>>>(thbf, ntbf, vsum, agg);
    int tpad = NB * HP * WP * C2;
    build_pad_kernel<<<(tpad + 255) / 256, 256, 0, stream>>>(agg, nodes_cl, xpad);
    conv3x3_kernel<C2, 0><<<dim3(10, 4, NB), 256, 0, stream>>>(
        xpad, wg_t, gates_b, nodes_cl, nullptr, rh, zbuf);
    build_pad_kernel<<<(tpad + 255) / 256, 256, 0, stream>>>(agg, rh, xpad);
    conv3x3_kernel<C_DIM, 1><<<dim3(10, 2, NB), 256, 0, stream>>>(
        xpad, wc_t, can_b, nodes_cl, zbuf, nodes_cl, nullptr);
  }
  from_cl_kernel<<<(ETOT + 255) / 256, 256, 0, stream>>>(nodes_cl, (float*)d_out);
}
